// MultiEmbodimentActionEncoder_15642270892656
// MI455X (gfx1250) — matrix operand order repaired and checked by reading, not compiled, not run on hardware
//
#include <hip/hip_runtime.h>

// ---------------------------------------------------------------------------
// MultiEmbodimentActionEncoder for MI455X (gfx1250), wave32 + WMMA f16.
//   L1: a_emb = actions @ W1[cat] + b1   (K=32)
//   x2  = concat(a_emb, tau)             (row stride 2048)
//   L2: h = silu(x2 @ W2[cat] + b2)      (K=2048)
//   L3: out = h @ W3[cat] + b3           (K=1024)
//
// Batches are grouped into supertiles of <=4 same-category batch elements so
// each block amortizes weight reads over M=256 rows (4x less HBM weight
// traffic vs per-batch tiling). GEMM core: v_wmma_f32_16x16x32_f16, f32 acc.
// LDS tiles use a 72-short (36-dword) row stride: 36*d % 64 == 0 only at
// d=16 -> conflict-free b128 fragment loads across the 16 lanes.
// ---------------------------------------------------------------------------

#define B_SZ  128
#define T_SZ  64
#define H_SZ  1024
#define KC    64      // K-chunk staged in LDS
#define KCP   72      // padded LDS row stride (shorts): 144B, 16B aligned
#define MT    256     // M rows per block (4 batches x 64)
#define NTB   64      // N cols per block
#define NSUP  56      // max supertiles: sum ceil(c_i/4) <= (128 + 3*32)/4

typedef __attribute__((ext_vector_type(16))) _Float16     v16h;
typedef __attribute__((ext_vector_type(8)))  float        v8f;
typedef __attribute__((ext_vector_type(4)))  unsigned int u32x4;
typedef __attribute__((ext_vector_type(2)))  unsigned int u32x2;
typedef __attribute__((ext_vector_type(4)))  float        f32x4;

union Frag {
  v16h  v;
  u32x4 q[2];
};

__device__ __forceinline__ unsigned short f2h(float f) {
  union { _Float16 h; unsigned short u; } c;
  c.h = (_Float16)f;
  return c.u;
}

// ---------------------------------------------------------------------------
// Kernel 0: deterministic counting-sort of batches by category into
// supertiles of up to 4 batches. sb[s][0..3] = batch ids (-1 pad), sc[s]=cat.
// ---------------------------------------------------------------------------
__global__ __launch_bounds__(128) void build_groups_kernel(
    const int* __restrict__ cat, int* __restrict__ sb, int* __restrict__ sc)
{
  __shared__ int lcat[B_SZ];
  __shared__ int nch[32];
  __shared__ int start[33];
  const int tid = threadIdx.x;

  if (tid < B_SZ) lcat[tid] = cat[tid];
  __syncthreads();

  if (tid < 32) {
    int cnt = 0;
    for (int b = 0; b < B_SZ; ++b) cnt += (lcat[b] == tid) ? 1 : 0;
    nch[tid] = (cnt + 3) >> 2;
  }
  __syncthreads();

  if (tid == 0) {
    int s = 0;
    for (int c = 0; c < 32; ++c) { start[c] = s; s += nch[c]; }
    start[32] = s;
  }
  __syncthreads();

  if (tid < 32) {
    int s = start[tid];
    int slot = 0;
    for (int b = 0; b < B_SZ; ++b) {
      if (lcat[b] == tid) {
        if (slot == 0) {
          sc[s] = tid;
          sb[s * 4 + 0] = sb[s * 4 + 1] = sb[s * 4 + 2] = sb[s * 4 + 3] = -1;
        }
        sb[s * 4 + slot] = b;
        if (++slot == 4) { slot = 0; ++s; }
      }
    }
  }
  __syncthreads();

  if (tid == 0) {
    for (int s = start[32]; s < NSUP; ++s) {
      sc[s] = 0;
      sb[s * 4 + 0] = sb[s * 4 + 1] = sb[s * 4 + 2] = sb[s * 4 + 3] = -1;
    }
  }
}

// ---------------------------------------------------------------------------
// Kernel 1: pack actions -> f16 (X1); write sinusoidal tau into second half
// of x2 [B][T][2048] f16. Grid: B*T*1024/256.
// ---------------------------------------------------------------------------
__global__ __launch_bounds__(256) void pack_tau_kernel(
    const float* __restrict__ actions,
    const float* __restrict__ timesteps,
    unsigned short* __restrict__ X1,
    unsigned short* __restrict__ X2)
{
  int idx = blockIdx.x * 256 + threadIdx.x;      // < 8388608
  int b = idx >> 16;
  int r = idx & 65535;
  int t = r >> 10;
  int j = r & 1023;

  float ts = timesteps[b];
  int   jj = (j < 512) ? j : (j - 512);
  float ex = __expf(-9.210340371976184f * (float)jj * (1.0f / 512.0f));
  float fr = ts * ex;
  float v  = (j < 512) ? __sinf(fr) : __cosf(fr);
  X2[((size_t)((b << 6) + t) << 11) + 1024 + j] = f2h(v);

  if (idx < B_SZ * T_SZ * 32) X1[idx] = f2h(actions[idx]);
}

// ---------------------------------------------------------------------------
// Kernel 2: supertile category GEMM + bias (+ SiLU).
// Block: 256 threads (8 waves): wave = (wm in 0..3 over 64-row quarters) x
// (wn in 0..1 over 32-col halves). Tile M=256, N=64, K-chunk 64.
// Grid: (N/64, NSUP).
// ---------------------------------------------------------------------------
__global__ __launch_bounds__(256) void gemm_cat_kernel(
    const unsigned short* __restrict__ A,    // [B][64][K] f16
    const float* __restrict__ W,             // [32][K][N] f32
    const float* __restrict__ bias,          // [32][N] f32
    const int* __restrict__ sb,              // [NSUP][4] batch ids
    const int* __restrict__ sc,              // [NSUP] category
    unsigned short* __restrict__ out_h,      // f16 out (or nullptr)
    float* __restrict__ out_f,               // f32 out (or nullptr)
    int K, int N, int ldo, int act)
{
  __shared__ unsigned short As[MT * KCP];    // 36 KB, [row][k], padded stride
  __shared__ unsigned short Bs[NTB * KCP];   // 9 KB, [n][k] transposed, padded

  const int s = blockIdx.y;
  int bq0 = sb[s * 4 + 0], bq1 = sb[s * 4 + 1];
  int bq2 = sb[s * 4 + 2], bq3 = sb[s * 4 + 3];
  if (bq0 < 0) return;                       // padded supertile: whole block exits

  const int n0   = blockIdx.x * NTB;
  const int c    = sc[s];
  const int tid  = threadIdx.x;
  const int wave = tid >> 5;
  const int wm   = wave >> 1;                // 0..3: 64-row quarter
  const int wn   = wave & 1;                 // 0..1: 32-col half
  const int lane = tid & 31;
  const int l15  = lane & 15;
  const int hsel = lane >> 4;

  const float* Wc = W + (size_t)c * (size_t)K * (size_t)N;

  v8f acc[4][2];
#pragma unroll
  for (int i = 0; i < 4; ++i)
#pragma unroll
    for (int j = 0; j < 2; ++j) acc[i][j] = (v8f)0.0f;

  for (int kc = 0; kc < K; kc += KC) {
    const bool full = (kc + KC) <= K;

    // ---- stage A: 256 rows x 64 k (f16). 2048 b128 loads, 8 per thread.
    // j even/odd pairs cover one 64-row quarter -> q = j>>1 is compile-time.
#pragma unroll
    for (int j = 0; j < 8; ++j) {
      const int q  = j >> 1;
      const int bq = (q == 0) ? bq0 : (q == 1) ? bq1 : (q == 2) ? bq2 : bq3;
      const int t  = (tid >> 3) + (j & 1) * 32;
      const int c8 = (tid & 7) * 8;
      const int k  = kc + c8;
      u32x4 v = (u32x4)0u;
      if (bq >= 0 && (full || k < K))
        v = *(const u32x4*)(A + ((size_t)bq * T_SZ + t) * K + k);
      *(u32x4*)(&As[(q * 64 + t) * KCP + c8]) = v;
    }

    // ---- stage W: 64 k x 64 n f32 -> f16 transposed into Bs[n][k].
    // Thread handles one column n, 4 consecutive k -> one ds_store_b64.
    {
      const int n = tid & 63;
#pragma unroll
      for (int j = 0; j < 4; ++j) {
        const int kg = (tid >> 6) + j * 4;     // 0..15
        const int k  = kc + kg * 4;
        float v0 = 0.f, v1 = 0.f, v2 = 0.f, v3 = 0.f;
        if (full) {
          const float* p = Wc + (size_t)k * N + n0 + n;
          v0 = p[0]; v1 = p[(size_t)N]; v2 = p[(size_t)2 * N]; v3 = p[(size_t)3 * N];
        } else {
          const float* p = Wc + (size_t)k * N + n0 + n;
          if (k + 0 < K) v0 = p[0];
          if (k + 1 < K) v1 = p[(size_t)N];
          if (k + 2 < K) v2 = p[(size_t)2 * N];
          if (k + 3 < K) v3 = p[(size_t)3 * N];
        }
        u32x2 pk;
        pk.x = (unsigned)f2h(v0) | ((unsigned)f2h(v1) << 16);
        pk.y = (unsigned)f2h(v2) | ((unsigned)f2h(v3) << 16);
        *(u32x2*)(&Bs[n * KCP + kg * 4]) = pk;
      }
    }

    // ---- prefetch next weight chunk into GL2 (global_prefetch_b8) ----
    if (full && (kc + KC) < K && tid < 128) {
      const int row = tid >> 1, hf = tid & 1;
      __builtin_prefetch(Wc + (size_t)(kc + KC + row) * N + n0 + hf * 32, 0, 3);
    }

    __syncthreads();

    // ---- WMMA: two 16x16x32 K-steps; 16 WMMAs per wave per chunk ----
#pragma unroll
    for (int kk = 0; kk < KC; kk += 32) {
      Frag a4[4];
#pragma unroll
      for (int mt = 0; mt < 4; ++mt) {
        const unsigned short* ap = &As[(wm * 64 + mt * 16 + l15) * KCP + kk + hsel * 8];
        a4[mt].q[0] = *(const u32x4*)(ap);
        a4[mt].q[1] = *(const u32x4*)(ap + 16);
      }
#pragma unroll
      for (int nt = 0; nt < 2; ++nt) {
        Frag bf;
        const unsigned short* bp = &Bs[(wn * 32 + nt * 16 + l15) * KCP + kk + hsel * 8];
        bf.q[0] = *(const u32x4*)(bp);
        bf.q[1] = *(const u32x4*)(bp + 16);
#pragma unroll
        for (int mt = 0; mt < 4; ++mt)
          acc[mt][nt] = __builtin_amdgcn_wmma_f32_16x16x32_f16(
              false, a4[mt].v, false, bf.v, (short)0, acc[mt][nt], false, false);
      }
    }
    __syncthreads();
  }

  // ---- epilogue: this wave's quarter is one batch element ----
  const int bep = sb[s * 4 + wm];
  if (bep >= 0) {
#pragma unroll
    for (int nt = 0; nt < 2; ++nt) {
      const int   col = n0 + wn * 32 + nt * 16 + l15;
      const float bv  = bias[(size_t)c * N + col];
#pragma unroll
      for (int mt = 0; mt < 4; ++mt) {
#pragma unroll
        for (int i = 0; i < 8; ++i) {
          const int t = mt * 16 + i + hsel * 8;      // row within batch, 0..63
          float x = acc[mt][nt][i] + bv;
          if (act) x = x / (1.0f + __expf(-x));
          const size_t off = ((size_t)(bep * T_SZ + t)) * (size_t)ldo + col;
          if (out_h) out_h[off] = f2h(x);
          else       out_f[off] = x;
        }
      }
    }
  }
}

// ---------------------------------------------------------------------------
// Launcher. inputs: actions, timesteps, W1_w, W1_b, W2_w, W2_b, W3_w, W3_b,
// category_ids
// ---------------------------------------------------------------------------
extern "C" void kernel_launch(void* const* d_in, const int* in_sizes, int n_in,
                              void* d_out, int out_size, void* d_ws, size_t ws_size,
                              hipStream_t stream) {
  (void)in_sizes; (void)n_in; (void)out_size; (void)ws_size;

  const float* actions   = (const float*)d_in[0];
  const float* timesteps = (const float*)d_in[1];
  const float* W1w       = (const float*)d_in[2];
  const float* W1b       = (const float*)d_in[3];
  const float* W2w       = (const float*)d_in[4];
  const float* W2b       = (const float*)d_in[5];
  const float* W3w       = (const float*)d_in[6];
  const float* W3b       = (const float*)d_in[7];
  const int*   cat       = (const int*)d_in[8];
  float*       out       = (float*)d_out;

  // ws layout: [sb 224 | sc 56 | pad -> 288 ints] [X1][X2][H] (f16 as ushort)
  int* sb = (int*)d_ws;
  int* sc = sb + NSUP * 4;
  unsigned short* X1 = (unsigned short*)(sb + 288);              // 1152B offset
  unsigned short* X2 = X1 + (size_t)B_SZ * T_SZ * 32;
  unsigned short* Hb = X2 + (size_t)B_SZ * T_SZ * 2 * H_SZ;

  build_groups_kernel<<<1, 128, 0, stream>>>(cat, sb, sc);

  pack_tau_kernel<<<dim3((B_SZ * T_SZ * H_SZ) / 256), 256, 0, stream>>>(
      actions, timesteps, X1, X2);

  dim3 grid(H_SZ / NTB, NSUP);

  // L1: a_emb -> first half of X2 (row stride 2048)
  gemm_cat_kernel<<<grid, 256, 0, stream>>>(
      X1, W1w, W1b, sb, sc, X2, nullptr, /*K=*/32, /*N=*/H_SZ, /*ldo=*/2 * H_SZ, /*act=*/0);

  // L2: silu(x2 @ W2 + b2) -> H (f16)
  gemm_cat_kernel<<<grid, 256, 0, stream>>>(
      X2, W2w, W2b, sb, sc, Hb, nullptr, /*K=*/2 * H_SZ, /*N=*/H_SZ, /*ldo=*/H_SZ, /*act=*/1);

  // L3: H @ W3 + b3 -> out (f32)
  gemm_cat_kernel<<<grid, 256, 0, stream>>>(
      Hb, W3w, W3b, sb, sc, nullptr, out, /*K=*/H_SZ, /*N=*/H_SZ, /*ldo=*/H_SZ, /*act=*/0);
}
